// KPlaneField_54322746360049
// MI455X (gfx1250) — matrix operand order repaired and checked by reading, not compiled, not run on hardware
//
#include <hip/hip_runtime.h>
#include <hip/hip_fp16.h>
#include <cstdint>
#include <cstddef>

#define N_RAYS    8192
#define N_SAMPLES 48
#define NPTS      (N_RAYS * N_SAMPLES)
#define NTILES    (NPTS / 16)
#define GEO       15

typedef __attribute__((ext_vector_type(16))) _Float16 v16h;
typedef __attribute__((ext_vector_type(8)))  _Float16 v8h;
typedef __attribute__((ext_vector_type(8)))  float    v8f;

#define WAIT_DS() asm volatile("s_wait_dscnt 0" ::: "memory")

struct GridPtrs { const float*  g[12]; };
struct TGPtrs   { const __half* t[12]; };

// ---------------- Pass A: [32][R][R] f32 -> [R*R][32] f16 ------------------
// LDS-tiled so both the f32 reads (coalesced along pixels per channel) and
// the f16 writes (16B v8h per lane, 512B per wave) are fully coalesced.
#define TPIX 256
__global__ __launch_bounds__(256) void kp_transpose(const float* __restrict__ src,
                                                    __half* __restrict__ dst, int R)
{
    __shared__ float tile[32][TPIX + 1];   // +1 pad: stride 257 -> conflict-free columns
    const size_t rr = (size_t)R * R;       // always a multiple of TPIX
    const int tid = threadIdx.x;
    for (size_t pb = (size_t)blockIdx.x * TPIX; pb < rr; pb += (size_t)gridDim.x * TPIX) {
#pragma unroll
        for (int c = 0; c < 32; ++c)
            tile[c][tid] = src[(size_t)c * rr + pb + tid];
        __syncthreads();
#pragma unroll
        for (int j = 0; j < 4; ++j) {
            int flat = j * 2048 + tid * 8;   // 0..8191, multiple of 8
            int pix  = flat >> 5;
            int c0   = flat & 31;            // 0, 8, 16, 24
            v8h v;
#pragma unroll
            for (int e = 0; e < 8; ++e) v[e] = (_Float16)tile[c0 + e][pix];
            *(v8h*)((_Float16*)dst + (pb + pix) * 32 + c0) = v;
        }
        __syncthreads();
    }
}

// ---------------- Pass B: pack W[N][K] f32 -> WMMA B-operand f16 -----------
// Tile (c,t): lane l (g=l>>4, n=16t+(l&15)) holds B[K=32c+(e<8 ? 8g+e : 16+8g+e-8)][n], e=0..15.
__global__ void kp_pack_w(const float* __restrict__ W, __half* __restrict__ dst,
                          int Ntot, int Ktot, int Nt)
{
    int tileid = blockIdx.x;            // c*Nt + t
    int t = tileid % Nt, c = tileid / Nt;
    int l = threadIdx.x;
    int g = l >> 4, n = 16 * t + (l & 15);
#pragma unroll
    for (int e = 0; e < 16; ++e) {
        int k = 32 * c + (e < 8 ? 8 * g + e : 16 + 8 * g + (e - 8));
        float v = (n < Ntot && k < Ktot) ? W[(size_t)n * Ktot + k] : 0.0f;
        dst[((size_t)tileid * 32 + l) * 16 + e] = __float2half(v);
    }
}

// ---------------- Pass C: featurize (wave = point, lane = channel) ---------
__global__ void kp_featurize(const float* __restrict__ pts, const float* __restrict__ aabb,
                             GridPtrs G, TGPtrs T, int use_tg, __half* __restrict__ feats)
{
    const int lane = threadIdx.x & 31;
    const int wave = threadIdx.x >> 5;
    const int wpb  = blockDim.x >> 5;
    const float lo0 = aabb[0], lo1 = aabb[1], lo2 = aabb[2];
    const float s0 = 2.0f / (aabb[3] - aabb[0]);
    const float s1 = 2.0f / (aabb[4] - aabb[1]);
    const float s2 = 2.0f / (aabb[5] - aabb[2]);
    const int axs[3] = {0, 0, 1};
    const int ays[3] = {1, 2, 2};

    for (size_t pt = (size_t)blockIdx.x * wpb + wave; pt < NPTS;
         pt += (size_t)gridDim.x * wpb) {
        float P[3];
        P[0] = (pts[pt * 3 + 0] - lo0) * s0 - 1.0f;
        P[1] = (pts[pt * 3 + 1] - lo1) * s1 - 1.0f;
        P[2] = (pts[pt * 3 + 2] - lo2) * s2 - 1.0f;
#pragma unroll
        for (int s = 0; s < 4; ++s) {
            const int R = 128 << s;
            float interp = 1.0f;
#pragma unroll
            for (int p = 0; p < 3; ++p) {
                float cx = P[axs[p]], cy = P[ays[p]];
                float fx = fminf(fmaxf((cx + 1.0f) * 0.5f * (float)(R - 1), 0.0f), (float)(R - 1));
                float fy = fminf(fmaxf((cy + 1.0f) * 0.5f * (float)(R - 1), 0.0f), (float)(R - 1));
                float x0f = floorf(fx), y0f = floorf(fy);
                int x0 = (int)x0f, y0 = (int)y0f;
                int x1 = min(x0 + 1, R - 1), y1 = min(y0 + 1, R - 1);
                float wx = fx - x0f, wy = fy - y0f;
                float v00, v01, v10, v11;
                const int gi = s * 3 + p;
                if (use_tg) {
                    const __half* b = T.t[gi];
                    v00 = __half2float(b[((size_t)y0 * R + x0) * 32 + lane]);
                    v01 = __half2float(b[((size_t)y0 * R + x1) * 32 + lane]);
                    v10 = __half2float(b[((size_t)y1 * R + x0) * 32 + lane]);
                    v11 = __half2float(b[((size_t)y1 * R + x1) * 32 + lane]);
                } else {
                    const float* b = G.g[gi] + (size_t)lane * R * R;
                    v00 = b[(size_t)y0 * R + x0];
                    v01 = b[(size_t)y0 * R + x1];
                    v10 = b[(size_t)y1 * R + x0];
                    v11 = b[(size_t)y1 * R + x1];
                }
                float top = v00 * (1.0f - wx) + v01 * wx;
                float bot = v10 * (1.0f - wx) + v11 * wx;
                interp *= top * (1.0f - wy) + bot * wy;
            }
            feats[pt * 128 + s * 32 + lane] = __float2half(interp);
        }
    }
}

// ---------------- Pass D helpers -------------------------------------------
// A-operand (16x32 f16): lane row = lane&15; elems 0..7 = K[8g..8g+7],
// elems 8..15 = K[16+8g .. 16+8g+7] (both contiguous 16B runs).
__device__ __forceinline__ v16h kp_load_a(const _Float16* rowbase, int g)
{
    v8h lo = *(const v8h*)(rowbase + 8 * g);
    v8h hi = *(const v8h*)(rowbase + 16 + 8 * g);
    v16h r;
#pragma unroll
    for (int i = 0; i < 8; ++i) { r[i] = lo[i]; r[8 + i] = hi[i]; }
    return r;
}

__device__ __forceinline__ v16h kp_load_b(const _Float16* wp, int tileid, int lane)
{
    return *(const v16h*)(wp + ((size_t)tileid * 32 + lane) * 16);
}

#define WMMA(a, b, c) \
    __builtin_amdgcn_wmma_f32_16x16x32_f16(false, (a), false, (b), (short)0, (c), false, false)

// weight-pack tile offsets
#define OFF_S0 0
#define OFF_S1 16
#define OFF_C0 18
#define OFF_C1 22
#define OFF_C2 30

// ---------------- Pass D: fused MLP, wave = 16-point tile ------------------
__global__ __launch_bounds__(256) void kp_mlp(const __half* __restrict__ feats_h,
                                              const __half* __restrict__ wpack_h,
                                              const float* __restrict__ dirs,
                                              float* __restrict__ out_rgb,
                                              float* __restrict__ out_den)
{
    __shared__ _Float16 sH [8][16 * 64];
    __shared__ _Float16 sCF[8][16 * 32];
    __shared__ float    sH1[8][16 * 16];

    const _Float16* feats = (const _Float16*)feats_h;
    const _Float16* wp    = (const _Float16*)wpack_h;
    const int lane = threadIdx.x & 31;
    const int w    = threadIdx.x >> 5;
    const int g    = lane >> 4;
    const int n    = lane & 15;
    _Float16* Hm = sH[w];
    _Float16* CF = sCF[w];
    float*    H1 = sH1[w];
    const v8f vzero = {0.f, 0.f, 0.f, 0.f, 0.f, 0.f, 0.f, 0.f};

    for (int tile = blockIdx.x * 8 + w; tile < NTILES; tile += gridDim.x * 8) {
        const size_t pt0 = (size_t)tile * 16;

        // ---- sigma L0: [16x128] @ [128x64], relu -> Hm ----
        // Issue all 8 A-loads (one clause) before the 16-WMMA burst.
        v16h a[4];
#pragma unroll
        for (int c = 0; c < 4; ++c)
            a[c] = kp_load_a(feats + (pt0 + n) * 128 + c * 32, g);
        v8f acc[4] = {vzero, vzero, vzero, vzero};
#pragma unroll
        for (int c = 0; c < 4; ++c)
#pragma unroll
            for (int t = 0; t < 4; ++t)
                acc[t] = WMMA(a[c], kp_load_b(wp, OFF_S0 + c * 4 + t, lane), acc[t]);
#pragma unroll
        for (int t = 0; t < 4; ++t)
#pragma unroll
            for (int i = 0; i < 8; ++i)
                Hm[(8 * g + i) * 64 + 16 * t + n] = (_Float16)fmaxf(acc[t][i], 0.0f);
        WAIT_DS();

        // ---- sigma L1: [16x64] @ [64x16] -> H1 (no relu) ----
        v8f h1 = vzero;
#pragma unroll
        for (int c = 0; c < 2; ++c)
            h1 = WMMA(kp_load_a(Hm + n * 64 + c * 32, g),
                      kp_load_b(wp, OFF_S1 + c, lane), h1);
#pragma unroll
        for (int i = 0; i < 8; ++i) H1[(8 * g + i) * 16 + n] = h1[i];
        WAIT_DS();

        // ---- SH encoding + cf = [enc16 | geo15 | 0], density = exp ----
        if (lane < 16) {
            const int r = lane;
            const size_t pt = pt0 + r;
            const int ray = (int)(pt / N_SAMPLES);
            const float x = dirs[ray * 3 + 0], y = dirs[ray * 3 + 1], z = dirs[ray * 3 + 2];
            const float x2 = x * x, y2 = y * y, z2 = z * z;
            const float xy = x * y, yz = y * z, xz = x * z;
            float sh[16];
            sh[0]  = 0.28209479177387814f;
            sh[1]  = -0.48860251190291987f * y;
            sh[2]  =  0.48860251190291987f * z;
            sh[3]  = -0.48860251190291987f * x;
            sh[4]  =  1.0925484305920792f * xy;
            sh[5]  = -1.0925484305920792f * yz;
            sh[6]  =  0.94617469575756f * z2 - 0.31539156525252f;
            sh[7]  = -1.0925484305920792f * xz;
            sh[8]  =  0.5462742152960396f * (x2 - y2);
            sh[9]  =  0.5900435899266435f * y * (-3.0f * x2 + y2);
            sh[10] =  2.890611442640554f * xy * z;
            sh[11] =  0.4570457994644657f * y * (1.0f - 5.0f * z2);
            sh[12] =  0.3731763325901154f * z * (5.0f * z2 - 3.0f);
            sh[13] =  0.4570457994644657f * x * (1.0f - 5.0f * z2);
            sh[14] =  1.445305697320134f * z * (x2 - y2);
            sh[15] =  0.5900435899266435f * x * (-x2 + 3.0f * y2);
#pragma unroll
            for (int j = 0; j < 16; ++j) CF[r * 32 + j] = (_Float16)sh[j];
#pragma unroll
            for (int j = 0; j < GEO; ++j) CF[r * 32 + 16 + j] = (_Float16)H1[r * 16 + j];
            CF[r * 32 + 31] = (_Float16)0.0f;
            out_den[pt] = __expf(H1[r * 16 + GEO]);
        }
        WAIT_DS();

        // ---- color L0: [16x32] @ [32x64], relu -> Hm ----
        v8f c0[4] = {vzero, vzero, vzero, vzero};
        {
            v16h ca = kp_load_a(CF + n * 32, g);
#pragma unroll
            for (int t = 0; t < 4; ++t)
                c0[t] = WMMA(ca, kp_load_b(wp, OFF_C0 + t, lane), c0[t]);
        }
#pragma unroll
        for (int t = 0; t < 4; ++t)
#pragma unroll
            for (int i = 0; i < 8; ++i)
                Hm[(8 * g + i) * 64 + 16 * t + n] = (_Float16)fmaxf(c0[t][i], 0.0f);
        WAIT_DS();

        // ---- color L1: [16x64] @ [64x64], relu -> Hm ----
        v8f c1[4] = {vzero, vzero, vzero, vzero};
        {
            v16h a0 = kp_load_a(Hm + n * 64 + 0,  g);
            v16h a1 = kp_load_a(Hm + n * 64 + 32, g);
#pragma unroll
            for (int t = 0; t < 4; ++t) {
                c1[t] = WMMA(a0, kp_load_b(wp, OFF_C1 + 0 * 4 + t, lane), c1[t]);
                c1[t] = WMMA(a1, kp_load_b(wp, OFF_C1 + 1 * 4 + t, lane), c1[t]);
            }
        }
        WAIT_DS();
#pragma unroll
        for (int t = 0; t < 4; ++t)
#pragma unroll
            for (int i = 0; i < 8; ++i)
                Hm[(8 * g + i) * 64 + 16 * t + n] = (_Float16)fmaxf(c1[t][i], 0.0f);
        WAIT_DS();

        // ---- color L2: [16x64] @ [64x16(pad of 3)], sigmoid -> rgb ----
        v8f o = vzero;
        o = WMMA(kp_load_a(Hm + n * 64 + 0,  g), kp_load_b(wp, OFF_C2 + 0, lane), o);
        o = WMMA(kp_load_a(Hm + n * 64 + 32, g), kp_load_b(wp, OFF_C2 + 1, lane), o);
        if (n < 3) {
#pragma unroll
            for (int i = 0; i < 8; ++i) {
                size_t pt = pt0 + 8 * g + i;
                out_rgb[pt * 3 + n] = 1.0f / (1.0f + __expf(-o[i]));
            }
        }
    }
}

// --------------------------------------------------------------------------
extern "C" void kernel_launch(void* const* d_in, const int* in_sizes, int n_in,
                              void* d_out, int out_size, void* d_ws, size_t ws_size,
                              hipStream_t stream)
{
    (void)in_sizes; (void)n_in; (void)out_size;
    const float* pts  = (const float*)d_in[0];
    const float* dirs = (const float*)d_in[1];
    const float* aabb = (const float*)d_in[2];
    const float* w_s0 = (const float*)d_in[15];
    const float* w_s1 = (const float*)d_in[16];
    const float* w_c0 = (const float*)d_in[17];
    const float* w_c1 = (const float*)d_in[18];
    const float* w_c2 = (const float*)d_in[19];

    char* ws = (char*)d_ws;
    const size_t featBytes = (size_t)NPTS * 128 * 2;    // f16 features
    const size_t wBytes    = (size_t)32 * 512 * 2;      // 32 packed weight tiles

    __half* feats = (__half*)ws;
    __half* wpack = (__half*)(ws + featBytes);
    __half* tg    = (__half*)(ws + featBytes + wBytes);

    size_t tg_off[12];
    size_t tgElems = 0;
    for (int s = 0; s < 4; ++s) {
        size_t R = (size_t)128 << s;
        for (int p = 0; p < 3; ++p) { tg_off[s * 3 + p] = tgElems; tgElems += 32 * R * R; }
    }
    const int use_tg = (ws_size >= featBytes + wBytes + tgElems * 2) ? 1 : 0;

    GridPtrs G;
    TGPtrs   T;
    for (int i = 0; i < 12; ++i) {
        G.g[i] = (const float*)d_in[3 + i];
        T.t[i] = use_tg ? (tg + tg_off[i]) : (const __half*)ws;
    }

    if (use_tg) {
        for (int i = 0; i < 12; ++i) {
            int R = 128 << (i / 3);
            size_t nblk = ((size_t)R * R) / TPIX;       // R^2 is a multiple of 256
            if (nblk > 4096) nblk = 4096;
            kp_transpose<<<(unsigned)nblk, 256, 0, stream>>>(G.g[i], tg + tg_off[i], R);
        }
    }

    // pack all MLP weights into f16 WMMA B-operand layout
    kp_pack_w<<<16, 32, 0, stream>>>(w_s0, wpack + OFF_S0 * 512, 64, 128, 4);
    kp_pack_w<<< 2, 32, 0, stream>>>(w_s1, wpack + OFF_S1 * 512, 16,  64, 1);
    kp_pack_w<<< 4, 32, 0, stream>>>(w_c0, wpack + OFF_C0 * 512, 64,  31, 4);
    kp_pack_w<<< 8, 32, 0, stream>>>(w_c1, wpack + OFF_C1 * 512, 64,  64, 4);
    kp_pack_w<<< 2, 32, 0, stream>>>(w_c2, wpack + OFF_C2 * 512,  3,  64, 1);

    kp_featurize<<<8192, 128, 0, stream>>>(pts, aabb, G, T, use_tg, feats);

    float* out_rgb = (float*)d_out;
    float* out_den = out_rgb + (size_t)NPTS * 3;
    kp_mlp<<<NTILES / 8, 256, 0, stream>>>(feats, wpack, dirs, out_rgb, out_den);
}
